// AttentionBlockSE3_67405216743684
// MI455X (gfx1250) — matrix operand order repaired and checked by reading, not compiled, not run on hardware
//
#include <hip/hip_runtime.h>
#include <hip/hip_bf16.h>

typedef __attribute__((ext_vector_type(16))) _Float16 v16h;
typedef __attribute__((ext_vector_type(8)))  float    v8f;

// ---- ordered-uint encoding of float so unsigned atomicMax == float max ----
__device__ __forceinline__ unsigned ordenc(float f) {
    unsigned u = __float_as_uint(f);
    return (u & 0x80000000u) ? ~u : (u | 0x80000000u);
}
__device__ __forceinline__ float orddec(unsigned e) {
    return __uint_as_float((e & 0x80000000u) ? (e & 0x7FFFFFFFu) : ~e);
}

__device__ __forceinline__ void load8(const float* __restrict__ p, float* r) {
    // p is 16B aligned by construction (all offsets are multiples of 8 floats
    // from 256B-aligned hipMalloc bases, except +8/+16-float offsets which are
    // still 32B aligned)
    const float4* p4 = reinterpret_cast<const float4*>(p);
    float4 u = p4[0], v = p4[1];
    r[0] = u.x; r[1] = u.y; r[2] = u.z; r[3] = u.w;
    r[4] = v.x; r[5] = v.y; r[6] = v.z; r[7] = v.w;
}

// Zero the accumulated output region and the segment-reduction scratch.
__global__ void se3_init_kernel(float* __restrict__ out01, unsigned* __restrict__ m_enc,
                                float* __restrict__ denom, int n_out, int n_nh) {
    int tid = blockIdx.x * blockDim.x + threadIdx.x;
    if (tid < n_out) out01[tid] = 0.0f;
    if (tid < n_nh) { m_enc[tid] = 0u; denom[tid] = 0.0f; }
}

// Pass 1: one wave handles a tile of 16 edges; per head, one
// v_wmma_f32_16x16x32_f16 computes K_tile x Q_tile^T; diagonal = logits.
__global__ void se3_logits_wmma_kernel(const float* __restrict__ key0,   // [E,64]
                                       const float* __restrict__ key1,   // [E,192]
                                       const float* __restrict__ q0,     // [N,64]
                                       const float* __restrict__ q1,     // [N,192]
                                       const int*   __restrict__ dst,    // [E]
                                       float* __restrict__ prelogits,    // [E,8] (output 3)
                                       float* __restrict__ ws_logits,    // [E,8] scratch
                                       unsigned* __restrict__ m_enc,     // [N,8]
                                       int E, int T) {
    int wave = blockIdx.x * (blockDim.x >> 5) + (threadIdx.x >> 5);
    if (wave >= T) return;                       // uniform per wave: EXEC stays all-1s
    int lane = threadIdx.x & 31;
    int m    = lane & 15;
    int hi   = lane >> 4;

    long tileBase = (long)wave * 16;
    long eRow = tileBase + m;                    // edge for A row m / B column m
    if (eRow >= E) eRow = E - 1;                 // clamp (E is a multiple of 16 anyway)
    int node = dst[eRow];

    const float* kb0 = key0 + eRow * 64;
    const float* kb1 = key1 + eRow * 192;
    const float* qb0 = q0 + (long)node * 64;
    const float* qb1 = q1 + (long)node * 192;

    for (int h = 0; h < 8; ++h) {
        // Head feature f(0..31): f<8 -> deg0[h*8+f]; f>=8 -> deg1[h*24 + (f-8)]
        // A layout (16-bit A 16x32): lanes 0-15 halves 0..7 = K0..7, 8..15 = K16..23
        //                            lanes16-31 halves 0..7 = K8..15, 8..15 = K24..31
        const float* pA0 = hi ? (kb1 + h * 24)      : (kb0 + h * 8);
        const float* pA1 = hi ? (kb1 + h * 24 + 16) : (kb1 + h * 24 + 8);
        // B layout (32x16): same K order as A (lanes 0-15: K0..7,K16..23; lanes 16-31: K8..15,K24..31)
        const float* pB0 = hi ? (qb1 + h * 24)      : (qb0 + h * 8);
        const float* pB1 = hi ? (qb1 + h * 24 + 16) : (qb1 + h * 24 + 8);

        float fa0[8], fa1[8], fb0[8], fb1[8];
        load8(pA0, fa0); load8(pA1, fa1);
        load8(pB0, fb0); load8(pB1, fb1);

        v16h a, b;
#pragma unroll
        for (int i = 0; i < 8; ++i) {
            a[i]     = (_Float16)fa0[i];
            a[i + 8] = (_Float16)fa1[i];
            b[i]     = (_Float16)fb0[i];
            b[i + 8] = (_Float16)fb1[i];
        }
        v8f c = {};
        c = __builtin_amdgcn_wmma_f32_16x16x32_f16(false, a, false, b,
                                                   (short)0, c, false, false);
        // Diagonal D[i,i]: i<8 -> lane i, reg i ; i=8+s -> lane 24+s, reg s
        float d = 0.0f;
#pragma unroll
        for (int r = 0; r < 8; ++r) {
            if (lane == r || lane == 24 + r) d = c[r];
        }
        bool valid = (lane < 8) || (lane >= 24);
        if (valid) {
            long e = tileBase + (lane & 15);
            if (e < E) {
                float logit = d * 0.0625f;       // 1/sqrt(256)
                prelogits[e * 8 + h] = logit;
                ws_logits[e * 8 + h] = logit;
                atomicMax(&m_enc[(long)node * 8 + h], ordenc(logit));
            }
        }
    }
}

// Pass 2: w = exp(l - m[dst]); denom[n,h] += w (atomic). In-place over ws.
__global__ void se3_expsum_kernel(float* __restrict__ ws_lw,           // [E,8] in: logits, out: w
                                  const unsigned* __restrict__ m_enc,  // [N,8]
                                  float* __restrict__ denom,           // [N,8]
                                  const int* __restrict__ dst, int total) {
    int tid = blockIdx.x * blockDim.x + threadIdx.x;
    if (tid >= total) return;
    int e = tid >> 3, h = tid & 7;
    int n = dst[e];
    float mm = orddec(m_enc[(long)n * 8 + h]);
    float w = __expf(ws_lw[tid] - mm);
    ws_lw[tid] = w;
    atomicAdd(&denom[(long)n * 8 + h], w);
}

// Pass 3: alpha = w / denom[dst]; scatter-add alpha * value into out0/out1.
__global__ void se3_agg_kernel(const float* __restrict__ ws_w,    // [E,8]
                               const float* __restrict__ denom,   // [N,8]
                               const int*   __restrict__ dst,
                               const float* __restrict__ value0,  // [E,64]
                               const float* __restrict__ value1,  // [E,192]
                               float* __restrict__ out0,          // [N,64]
                               float* __restrict__ out1,          // [N,192]
                               int total /* E*64 */) {
    int tid = blockIdx.x * blockDim.x + threadIdx.x;
    if (tid >= total) return;
    long e = tid >> 6;
    int  c = tid & 63;
    int  h = c >> 3;
    int  n = dst[e];
    float alpha = ws_w[e * 8 + h] / denom[(long)n * 8 + h];
    atomicAdd(&out0[(long)n * 64 + c], alpha * value0[e * 64 + c]);
    long vb = (e * 64 + c) * 3;
    long ob = ((long)n * 64 + c) * 3;
#pragma unroll
    for (int d = 0; d < 3; ++d)
        atomicAdd(&out1[ob + d], alpha * value1[vb + d]);
}

extern "C" void kernel_launch(void* const* d_in, const int* in_sizes, int n_in,
                              void* d_out, int out_size, void* d_ws, size_t ws_size,
                              hipStream_t stream) {
    const float* value0 = (const float*)d_in[0];   // [E,64,1]
    const float* value1 = (const float*)d_in[1];   // [E,64,3]
    const float* key0   = (const float*)d_in[2];   // [E,64,1]
    const float* key1   = (const float*)d_in[3];   // [E,64,3]
    const float* q0     = (const float*)d_in[4];   // [N,64,1]
    const float* q1     = (const float*)d_in[5];   // [N,64,3]
    const int*   dst    = (const int*)d_in[6];     // [E]

    const int E = in_sizes[0] / 64;
    const int N = in_sizes[4] / 64;

    // d_out layout: out0 [N*64] | out1 [N*192] | prelogits [E*8]
    float* out0 = (float*)d_out;
    float* out1 = out0 + (size_t)N * 64;
    float* prel = out1 + (size_t)N * 192;

    // workspace: logits/w [E*8] | m_enc [N*8] | denom [N*8]
    float*    ws_lw  = (float*)d_ws;
    unsigned* m_enc  = (unsigned*)(ws_lw + (size_t)E * 8);
    float*    denom  = (float*)(m_enc + (size_t)N * 8);

    const int BLK = 256;

    // 1) zero accumulators
    {
        int n_out = N * 256;     // out0 + out1 floats
        int n_nh  = N * 8;
        int grid  = (n_out + BLK - 1) / BLK;
        se3_init_kernel<<<grid, BLK, 0, stream>>>(out0, m_enc, denom, n_out, n_nh);
    }
    // 2) WMMA logits + segment max
    {
        int T = (E + 15) / 16;                 // 16-edge tiles, one wave each
        int grid = (T + 7) / 8;                // 8 waves per 256-thread block
        se3_logits_wmma_kernel<<<grid, BLK, 0, stream>>>(key0, key1, q0, q1, dst,
                                                         prel, ws_lw, m_enc, E, T);
    }
    // 3) exp + segment sum
    {
        int total = E * 8;
        int grid = (total + BLK - 1) / BLK;
        se3_expsum_kernel<<<grid, BLK, 0, stream>>>(ws_lw, m_enc, denom, dst, total);
    }
    // 4) normalize + weighted scatter aggregation
    {
        int total = E * 64;
        int grid = (total + BLK - 1) / BLK;
        se3_agg_kernel<<<grid, BLK, 0, stream>>>(ws_lw, denom, dst, value0, value1,
                                                 out0, out1, total);
    }
}
